// PETER_39092792328997
// MI455X (gfx1250) — compile-verified, matrix operand order checked by reading, not run on hardware
//
#include <hip/hip_runtime.h>
#include <hip/hip_bf16.h>
#include <stdint.h>

// ---------------- model constants ----------------
#define Lc   131          // IOS + CW = 3 + 128
#define Bc   32
#define Ec   512
#define Hc   8
#define Dc   64
#define NHIDc 2048
#define NTOKc 20004
#define MTOK (Lc*Bc)      // 4192, multiple of 16
#define MVOC (129*Bc)     // 4128, multiple of 16

typedef _Float16 h8v  __attribute__((ext_vector_type(8)));
typedef _Float16 h16v __attribute__((ext_vector_type(16)));
typedef float    f8v  __attribute__((ext_vector_type(8)));

__device__ __forceinline__ h16v cat8(h8v lo, h8v hi) {
  return __builtin_shufflevector(lo, hi, 0,1,2,3,4,5,6,7,8,9,10,11,12,13,14,15);
}
__device__ __forceinline__ f8v wmma16x16x32(h16v a, h16v b, f8v c) {
  // D = A(16x32 f16) * B(32x16 f16) + C(f32)
  return __builtin_amdgcn_wmma_f32_16x16x32_f16(false, a, false, b, (short)0, c, false, false);
}

// ---- CDNA5 async copy to LDS (ASYNCcnt-tracked; ISA 15.18.3 op 98) ----
__device__ __forceinline__ void async_ld_lds_b128(uint32_t ldsoff, const void* gaddr) {
  asm volatile("global_load_async_to_lds_b128 %0, %1, off"
               :: "v"(ldsoff), "v"(gaddr) : "memory");
}
__device__ __forceinline__ void wait_async0() {
#if __has_builtin(__builtin_amdgcn_s_wait_asynccnt)
  __builtin_amdgcn_s_wait_asynccnt(0);
#else
  asm volatile("s_wait_asynccnt 0x0" ::: "memory");
#endif
}

// ---------------- fp32 -> fp16 weight conversion ----------------
__global__ __launch_bounds__(256) void f32_to_f16_kernel(const float* __restrict__ s,
                                                         _Float16* __restrict__ d, int n) {
  int i = blockIdx.x * 256 + threadIdx.x;
  if (i < n) d[i] = (_Float16)s[i];
}

// ---------------- build src: embeddings * sqrt(E) + positional encoding ----------------
__global__ __launch_bounds__(256) void build_src_kernel(const int* __restrict__ user,
                                                        const int* __restrict__ item,
                                                        const float* __restrict__ rating,
                                                        const int* __restrict__ text,
                                                        const float* __restrict__ uemb,
                                                        const float* __restrict__ iemb,
                                                        const float* __restrict__ temb,
                                                        float* __restrict__ x) {
  int idx = blockIdx.x * 256 + threadIdx.x;
  const int total = Lc * Bc * Ec;
  if (idx >= total) return;
  int e = idx & (Ec - 1);
  int t = idx >> 9;          // token index l*B + b
  int b = t & (Bc - 1);
  int l = t >> 5;
  float v;
  if      (l == 0) v = uemb[(size_t)user[b] * Ec + e];
  else if (l == 1) v = iemb[(size_t)item[b] * Ec + e];
  else if (l == 2) v = rating[b];
  else             v = temb[(size_t)text[(l - 3) * Bc + b] * Ec + e];
  int e2 = e & ~1;
  float div = __expf(-(float)e2 * (9.210340371976184f / 512.0f)); // ln(10000)/d
  float ang = (float)l * div;
  float pe = (e & 1) ? __cosf(ang) : __sinf(ang);
  x[idx] = v * 22.62741699796952f + pe;   // sqrt(512)
}

// ---------------- rating MLP: sigmoid(cat(u,i) @ W1^T + b1) @ W2^T + b2 ----------------
__global__ __launch_bounds__(256) void rating_mlp_kernel(const int* __restrict__ user,
                                                         const int* __restrict__ item,
                                                         const float* __restrict__ uemb,
                                                         const float* __restrict__ iemb,
                                                         const float* __restrict__ w1,
                                                         const float* __restrict__ b1,
                                                         const float* __restrict__ w2,
                                                         const float* __restrict__ b2,
                                                         float* __restrict__ out) {
  int b = blockIdx.x;
  int j = threadIdx.x;
  const float* ue = uemb + (size_t)user[b] * Ec;
  const float* ie = iemb + (size_t)item[b] * Ec;
  float part = 0.f;
#pragma unroll
  for (int rep = 0; rep < 2; ++rep) {
    int o = j + rep * 256;
    const float* wr = w1 + (size_t)o * (2 * Ec);
    float z = b1[o];
    for (int t = 0; t < Ec; ++t) z += wr[t] * ue[t];
    for (int t = 0; t < Ec; ++t) z += wr[Ec + t] * ie[t];
    float hs = 1.f / (1.f + __expf(-z));
    part += hs * w2[o];
  }
  __shared__ float red[256];
  red[j] = part; __syncthreads();
  for (int s = 128; s; s >>= 1) { if (j < s) red[j] += red[j + s]; __syncthreads(); }
  if (j == 0) out[b] = red[0] + b2[0];
}

// ---------------- generic WMMA GEMM: D[M,N] = A[M,K] @ W[N,K]^T + bias ----------------
// block = 256 threads (8 waves). Block tile: 16 rows x 512 cols.
// Each wave: 16x64 strip -> 4 accumulators, A fragment reused 4x per K-step.
// B row pointers clamped once (no divergence in the K-loop); stores guarded.
// A staged through LDS; for A16 the stage uses global_load_async_to_lds_b128.
template <bool A16, bool OUT16, bool RELU, bool VOCAB>
__global__ __launch_bounds__(256) void gemm_wmma_kernel(const void* __restrict__ Aptr,
                                                        const _Float16* __restrict__ W,
                                                        const float* __restrict__ bias,
                                                        void* __restrict__ Dptr,
                                                        int M, int N, int K) {
  __shared__ __align__(16) _Float16 sA[16 * 32];
  const int m0 = blockIdx.x * 16;
  const int wave = threadIdx.x >> 5;
  const int lane = threadIdx.x & 31;
  const int n0 = blockIdx.y * 512 + wave * 64;
  const int rr = lane & 15;
  const int koff = (lane >> 4) * 8;

  // clamped B row pointers (always legal loads; epilogue guards stores)
  const _Float16* Wr[4];
#pragma unroll
  for (int t = 0; t < 4; ++t) {
    int n = n0 + 16 * t + rr;
    int nc = (n < N) ? n : (N - 1);
    Wr[t] = W + (size_t)nc * K;
  }

  f8v acc[4] = {};
  for (int k0 = 0; k0 < K; k0 += 32) {
    // ---- cooperative A stage: 16 rows x 32 f16 cols ----
    if (A16) {
      // raw fp16 tile copy: CDNA5 async DMA into LDS, 16B per lane
      if (threadIdx.x < 64) {
        int r  = threadIdx.x >> 2;
        int c8 = (threadIdx.x & 3) * 8;
        const _Float16* gp = &((const _Float16*)Aptr)[(size_t)(m0 + r) * K + k0 + c8];
        async_ld_lds_b128((uint32_t)(uintptr_t)&sA[r * 32 + c8], (const void*)gp);
      }
      wait_async0();
    } else {
      int r  = threadIdx.x >> 4;
      int c2 = (threadIdx.x & 15) * 2;
      const float* Af = (const float*)Aptr;
      float2 v = *(const float2*)&Af[(size_t)(m0 + r) * K + k0 + c2];
      sA[r * 32 + c2]     = (_Float16)v.x;
      sA[r * 32 + c2 + 1] = (_Float16)v.y;
    }
    __syncthreads();

    h8v alo = *(const h8v*)&sA[rr * 32 + koff];
    h8v ahi = *(const h8v*)&sA[rr * 32 + 16 + koff];
    h16v a = cat8(alo, ahi);

#pragma unroll
    for (int t = 0; t < 4; ++t) {
      h8v blo = *(const h8v*)&Wr[t][k0 + koff];
      h8v bhi = *(const h8v*)&Wr[t][k0 + 16 + koff];
      acc[t] = wmma16x16x32(a, cat8(blo, bhi), acc[t]);
    }
    if (k0 + 32 < K) __builtin_prefetch((const void*)(Wr[0] + k0 + 32), 0, 1);
    __syncthreads();
  }

  const int mofs = (lane >> 4) * 8;
#pragma unroll
  for (int t = 0; t < 4; ++t) {
    int n = n0 + 16 * t + rr;
    if (n < N) {
      float bv = bias[n];
#pragma unroll
      for (int i = 0; i < 8; ++i) {
        int m = m0 + mofs + i;
        float v = acc[t][i] + bv;
        if (RELU) v = fmaxf(v, 0.f);
        int drow = VOCAB ? ((m < 32) ? (4096 + m) : (m - 32)) : m;
        if (OUT16) ((_Float16*)Dptr)[(size_t)drow * N + n] = (_Float16)v;
        else       ((float*)Dptr)[(size_t)drow * N + n] = v;
      }
    }
  }
}

// ---------------- attention scores: S = Q K^T / 8, PETER mask ----------------
// one wave per 16x16 tile; grid = (81 tiles, B*H)
__global__ __launch_bounds__(32) void attn_scores_kernel(const _Float16* __restrict__ qkv,
                                                         float* __restrict__ probs) {
  const int tile = blockIdx.x;
  const int mt = tile / 9, nt = tile % 9;
  const int b = blockIdx.y >> 3, h = blockIdx.y & 7;
  const int lane = threadIdx.x;
  const int rr = lane & 15, hf = lane >> 4, koff = hf * 8;
  const int m0 = mt * 16, n0 = nt * 16;

  // clamp OOB rows for loads (those outputs are never stored)
  const int mrow = (m0 + rr < Lc) ? (m0 + rr) : (Lc - 1);
  const int nrow = (n0 + rr < Lc) ? (n0 + rr) : (Lc - 1);
  const _Float16* qp = qkv + ((size_t)mrow * Bc + b) * (3 * Ec) + h * Dc + koff;
  const _Float16* kp = qkv + ((size_t)nrow * Bc + b) * (3 * Ec) + Ec + h * Dc + koff;

  f8v acc = {};
#pragma unroll
  for (int k0 = 0; k0 < Dc; k0 += 32) {
    h8v alo = *(const h8v*)(qp + k0);
    h8v ahi = *(const h8v*)(qp + k0 + 16);
    h8v blo = *(const h8v*)(kp + k0);
    h8v bhi = *(const h8v*)(kp + k0 + 16);
    acc = wmma16x16x32(cat8(alo, ahi), cat8(blo, bhi), acc);
  }

  const int n = n0 + rr;
#pragma unroll
  for (int i = 0; i < 8; ++i) {
    int m = m0 + hf * 8 + i;
    if (m < Lc && n < Lc) {
      float s = acc[i] * 0.125f;                     // 1/sqrt(64)
      bool vis = (n < 3) || (m >= 3 && n <= m);      // PETER mask
      probs[(((size_t)b * Hc + h) * Lc + m) * Lc + n] = vis ? s : -1e30f;
    }
  }
}

// ---------------- softmax over rows of [B,H,L,L], in place ----------------
__global__ __launch_bounds__(128) void softmax_rows_kernel(float* __restrict__ probs) {
  int row = blockIdx.x * 4 + (threadIdx.x >> 5);
  int lane = threadIdx.x & 31;
  float* p = probs + (size_t)row * Lc;
  float v[5];
  float mx = -1e38f;
#pragma unroll
  for (int j = 0; j < 5; ++j) {
    int c = lane + j * 32;
    v[j] = (c < Lc) ? p[c] : -1e38f;
    mx = fmaxf(mx, v[j]);
  }
  for (int off = 16; off; off >>= 1) mx = fmaxf(mx, __shfl_xor(mx, off, 32));
  float s = 0.f;
#pragma unroll
  for (int j = 0; j < 5; ++j) {
    int c = lane + j * 32;
    if (c < Lc) { v[j] = __expf(v[j] - mx); s += v[j]; }
  }
  for (int off = 16; off; off >>= 1) s += __shfl_xor(s, off, 32);
  float inv = 1.f / s;
#pragma unroll
  for (int j = 0; j < 5; ++j) {
    int c = lane + j * 32;
    if (c < Lc) p[c] = v[j] * inv;
  }
}

// ---------------- head-average attention weights -> d_out attns region ----------------
__global__ __launch_bounds__(256) void head_avg_kernel(const float* __restrict__ probs,
                                                       float* __restrict__ out) {
  int idx = blockIdx.x * 256 + threadIdx.x;
  const int total = Bc * Lc * Lc;
  if (idx >= total) return;
  int b = idx / (Lc * Lc), r = idx % (Lc * Lc);
  float s = 0.f;
#pragma unroll
  for (int h = 0; h < Hc; ++h) s += probs[((size_t)b * Hc + h) * (Lc * Lc) + r];
  out[idx] = s * 0.125f;
}

// ---------------- context = P @ V (one wave per 16x16 tile) ----------------
__global__ __launch_bounds__(32) void attn_apply_kernel(const float* __restrict__ probs,
                                                        const _Float16* __restrict__ qkv,
                                                        float* __restrict__ ctx) {
  const int tile = blockIdx.x;               // 36 = 9 m-tiles x 4 n-tiles
  const int mt = tile >> 2, nt = tile & 3;
  const int b = blockIdx.y >> 3, h = blockIdx.y & 7;
  const int lane = threadIdx.x;
  const int rr = lane & 15, hf = lane >> 4, koff = hf * 8;
  const int m0 = mt * 16, n0 = nt * 16;
  const float* P = probs + ((size_t)b * Hc + h) * (Lc * Lc);
  const int n = n0 + rr;

  f8v acc = {};
  for (int k0 = 0; k0 < 160; k0 += 32) {
    // A = P[m, k] (reduction dim edge must be zero-filled, not clamped)
    int m = m0 + rr;
    h8v alo = {}, ahi = {};
    if (m < Lc) {
      const float* prow = P + (size_t)m * Lc;
#pragma unroll
      for (int j = 0; j < 8; ++j) {
        int k1 = k0 + koff + j;
        int k2 = k1 + 16;
        alo[j] = (k1 < Lc) ? (_Float16)prow[k1] : (_Float16)0.f;
        ahi[j] = (k2 < Lc) ? (_Float16)prow[k2] : (_Float16)0.f;
      }
    }
    h8v blo, bhi;
#pragma unroll
    for (int j = 0; j < 8; ++j) {
      int k1 = k0 + koff + j;
      int k2 = k1 + 16;
      blo[j] = (k1 < Lc) ? qkv[((size_t)k1 * Bc + b) * (3 * Ec) + 2 * Ec + h * Dc + n]
                         : (_Float16)0.f;
      bhi[j] = (k2 < Lc) ? qkv[((size_t)k2 * Bc + b) * (3 * Ec) + 2 * Ec + h * Dc + n]
                         : (_Float16)0.f;
    }
    acc = wmma16x16x32(cat8(alo, ahi), cat8(blo, bhi), acc);
  }
#pragma unroll
  for (int i = 0; i < 8; ++i) {
    int m = m0 + hf * 8 + i;
    if (m < Lc) ctx[((size_t)m * Bc + b) * Ec + h * Dc + n] = acc[i];
  }
}

// ---------------- x = LayerNorm(x + d) * g + b, one wave per token row ----------------
__global__ __launch_bounds__(256) void add_ln_kernel(float* __restrict__ x,
                                                     const float* __restrict__ d,
                                                     const float* __restrict__ g,
                                                     const float* __restrict__ bb) {
  int row = blockIdx.x * 8 + (threadIdx.x >> 5);
  int lane = threadIdx.x & 31;
  size_t base = (size_t)row * Ec;
  float y[16];
  float s = 0.f, ss = 0.f;
#pragma unroll
  for (int j = 0; j < 16; ++j) {
    int e = lane + j * 32;
    float v = x[base + e] + d[base + e];
    y[j] = v; s += v; ss += v * v;
  }
  for (int off = 16; off; off >>= 1) {
    s  += __shfl_xor(s, off, 32);
    ss += __shfl_xor(ss, off, 32);
  }
  float mu = s * (1.f / Ec);
  float var = ss * (1.f / Ec) - mu * mu;
  float rstd = rsqrtf(var + 1e-5f);
#pragma unroll
  for (int j = 0; j < 16; ++j) {
    int e = lane + j * 32;
    x[base + e] = (y[j] - mu) * rstd * g[e] + bb[e];
  }
}

// ---------------- in-place log_softmax over rows of 20004 in d_out ----------------
__global__ __launch_bounds__(256) void log_softmax_kernel(float* __restrict__ data, int ncols) {
  float* row = data + (size_t)blockIdx.x * ncols;
  __shared__ float red[256];
  int t = threadIdx.x;
  float mx = -1e38f;
  for (int c = t; c < ncols; c += 256) mx = fmaxf(mx, row[c]);
  red[t] = mx; __syncthreads();
  for (int s = 128; s; s >>= 1) { if (t < s) red[t] = fmaxf(red[t], red[t + s]); __syncthreads(); }
  mx = red[0]; __syncthreads();
  float sum = 0.f;
  for (int c = t; c < ncols; c += 256) sum += __expf(row[c] - mx);
  red[t] = sum; __syncthreads();
  for (int s = 128; s; s >>= 1) { if (t < s) red[t] += red[t + s]; __syncthreads(); }
  float lse = mx + __logf(red[0]);
  __syncthreads();
  for (int c = t; c < ncols; c += 256) row[c] = row[c] - lse;
}

// ==================================================================================
extern "C" void kernel_launch(void* const* d_in, const int* in_sizes, int n_in,
                              void* d_out, int out_size, void* d_ws, size_t ws_size,
                              hipStream_t stream) {
  (void)in_sizes; (void)n_in; (void)out_size; (void)ws_size;

  const int*   user   = (const int*)  d_in[0];
  const int*   item   = (const int*)  d_in[1];
  const float* rating = (const float*)d_in[2];
  const int*   text   = (const int*)  d_in[3];
  const float* uemb   = (const float*)d_in[4];
  const float* iemb   = (const float*)d_in[5];
  const float* temb   = (const float*)d_in[6];
  const float* h2t_w  = (const float*)d_in[7];
  const float* h2t_b  = (const float*)d_in[8];
  const float* mlp_w1 = (const float*)d_in[9];
  const float* mlp_b1 = (const float*)d_in[10];
  const float* mlp_w2 = (const float*)d_in[11];
  const float* mlp_b2 = (const float*)d_in[12];
  const float* inp_w  = (const float*)d_in[13];
  const float* inp_b  = (const float*)d_in[14];
  const float* outp_w = (const float*)d_in[15];
  const float* outp_b = (const float*)d_in[16];
  const float* ff1_w  = (const float*)d_in[17];
  const float* ff1_b  = (const float*)d_in[18];
  const float* ff2_w  = (const float*)d_in[19];
  const float* ff2_b  = (const float*)d_in[20];
  const float* ln1_g  = (const float*)d_in[21];
  const float* ln1_b  = (const float*)d_in[22];
  const float* ln2_g  = (const float*)d_in[23];
  const float* ln2_b  = (const float*)d_in[24];

  // ---- workspace layout ----
  char* ws = (char*)d_ws;
  size_t off = 0;
  auto carve = [&](size_t bytes) { size_t o = off; off += (bytes + 255) & ~(size_t)255; return o; };
  _Float16* wInP  = (_Float16*)(ws + carve((size_t)2 * 1536 * 512 * 2));
  _Float16* wOutP = (_Float16*)(ws + carve((size_t)2 * 512 * 512 * 2));
  _Float16* wF1   = (_Float16*)(ws + carve((size_t)2 * 2048 * 512 * 2));
  _Float16* wF2   = (_Float16*)(ws + carve((size_t)2 * 512 * 2048 * 2));
  _Float16* wH2T  = (_Float16*)(ws + carve((size_t)NTOKc * 512 * 2));
  float*    x     = (float*)   (ws + carve((size_t)MTOK * Ec * 4));
  float*    tmp   = (float*)   (ws + carve((size_t)MTOK * Ec * 4));
  float*    ctx   = (float*)   (ws + carve((size_t)MTOK * Ec * 4));
  _Float16* qkv   = (_Float16*)(ws + carve((size_t)MTOK * 3 * Ec * 2));
  float*    probs = (float*)   (ws + carve((size_t)Bc * Hc * Lc * Lc * 4));
  _Float16* hid   = (_Float16*)(ws + carve((size_t)MTOK * NHIDc * 2));

  // ---- d_out layout: [log_word_prob | log_context_dis | rating | attns] ----
  float* out = (float*)d_out;
  const size_t WP_ROWS = 4096;                 // 128*32
  const size_t VOC_ROWS = WP_ROWS + 32;        // + context rows -> 4128 contiguous rows
  float* out_rating = out + (size_t)VOC_ROWS * NTOKc;           // 82,576,512
  float* out_attns  = out_rating + 32;                          // 82,576,544

  // ---- 1. convert weights to f16 ----
  auto conv = [&](const float* s, _Float16* d, int n) {
    f32_to_f16_kernel<<<(n + 255) / 256, 256, 0, stream>>>(s, d, n);
  };
  conv(inp_w,  wInP,  2 * 1536 * 512);
  conv(outp_w, wOutP, 2 * 512 * 512);
  conv(ff1_w,  wF1,   2 * 2048 * 512);
  conv(ff2_w,  wF2,   2 * 512 * 2048);
  conv(h2t_w,  wH2T,  NTOKc * 512);

  // ---- 2. build src / 3. rating MLP ----
  build_src_kernel<<<(Lc * Bc * Ec + 255) / 256, 256, 0, stream>>>(
      user, item, rating, text, uemb, iemb, temb, x);
  rating_mlp_kernel<<<Bc, 256, 0, stream>>>(user, item, uemb, iemb,
                                            mlp_w1, mlp_b1, mlp_w2, mlp_b2, out_rating);

  // ---- 4. transformer layers ----
  for (int l = 0; l < 2; ++l) {
    // QKV: [4192,512] @ [1536,512]^T -> f16 [4192,1536]
    gemm_wmma_kernel<false, true, false, false>
        <<<dim3(MTOK / 16, 1536 / 512), 256, 0, stream>>>(
            (const void*)x, wInP + (size_t)l * 1536 * 512, inp_b + (size_t)l * 1536,
            (void*)qkv, MTOK, 3 * Ec, Ec);

    // scores + softmax + head average
    attn_scores_kernel<<<dim3(81, Bc * Hc), 32, 0, stream>>>(qkv, probs);
    softmax_rows_kernel<<<(Bc * Hc * Lc) / 4, 128, 0, stream>>>(probs);
    head_avg_kernel<<<(Bc * Lc * Lc + 255) / 256, 256, 0, stream>>>(
        probs, out_attns + (size_t)l * Bc * Lc * Lc);

    // context = P @ V
    attn_apply_kernel<<<dim3(36, Bc * Hc), 32, 0, stream>>>(probs, qkv, ctx);

    // out-proj: [4192,512] @ [512,512]^T -> f32 tmp
    gemm_wmma_kernel<false, false, false, false>
        <<<dim3(MTOK / 16, 1), 256, 0, stream>>>(
            (const void*)ctx, wOutP + (size_t)l * 512 * 512, outp_b + (size_t)l * Ec,
            (void*)tmp, MTOK, Ec, Ec);
    add_ln_kernel<<<MTOK / 8, 256, 0, stream>>>(x, tmp, ln1_g + (size_t)l * Ec,
                                                ln1_b + (size_t)l * Ec);

    // FF1 (+ReLU) -> f16 hidden [4192,2048]
    gemm_wmma_kernel<false, true, true, false>
        <<<dim3(MTOK / 16, NHIDc / 512), 256, 0, stream>>>(
            (const void*)x, wF1 + (size_t)l * 2048 * 512, ff1_b + (size_t)l * NHIDc,
            (void*)hid, MTOK, NHIDc, Ec);

    // FF2: f16 A [4192,2048] @ [512,2048]^T -> f32 tmp (A staged via async LDS DMA)
    gemm_wmma_kernel<true, false, false, false>
        <<<dim3(MTOK / 16, 1), 256, 0, stream>>>(
            (const void*)hid, wF2 + (size_t)l * 512 * 2048, ff2_b + (size_t)l * Ec,
            (void*)tmp, MTOK, Ec, NHIDc);
    add_ln_kernel<<<MTOK / 8, 256, 0, stream>>>(x, tmp, ln2_g + (size_t)l * Ec,
                                                ln2_b + (size_t)l * Ec);
  }

  // ---- 5. vocab projection: rows x[2:] -> logits straight into d_out (remapped) ----
  gemm_wmma_kernel<false, false, false, true>
      <<<dim3(MVOC / 16, (NTOKc + 511) / 512), 256, 0, stream>>>(
          (const void*)(x + (size_t)2 * Bc * Ec), wH2T, h2t_b,
          (void*)out, MVOC, NTOKc, Ec);

  // ---- 6. in-place log_softmax over all 4128 vocab rows ----
  log_softmax_kernel<<<(int)VOC_ROWS, 256, 0, stream>>>(out, NTOKc);
}
